// MultiHeadAttention_46093589021334
// MI455X (gfx1250) — matrix operand order repaired and checked by reading, not compiled, not run on hardware
//
#include <hip/hip_runtime.h>
#include <hip/hip_bf16.h>

typedef _Float16 half_t;
typedef __attribute__((ext_vector_type(16))) _Float16 v16h;
typedef __attribute__((ext_vector_type(8)))  _Float16 v8h;
typedef __attribute__((ext_vector_type(8)))  float    v8f;
typedef __attribute__((ext_vector_type(4)))  unsigned u32x4;
typedef __attribute__((ext_vector_type(8)))  unsigned u32x8;

union V16U { v16h v; v8h h8[2]; };

#define WMMA_F16(a,b,c) __builtin_amdgcn_wmma_f32_16x16x32_f16(false,(a),false,(b),(short)0,(c),false,false)
#define SCALE_F (1.0f / 1073741824.0f)   /* 1 / 64^5 */

// ---- CDNA5 async global->LDS copy (16B per lane), tracked by ASYNCcnt ----
__device__ inline void async_copy16(unsigned lds_off, const half_t* gptr) {
    asm volatile("global_load_async_to_lds_b128 %0, %1, off"
                 :: "v"(lds_off), "v"((unsigned long long)(uintptr_t)gptr)
                 : "memory");
}
__device__ inline void async_wait0() {
    asm volatile("s_wait_asynccnt 0" ::: "memory");
}

// ---- CDNA5 Tensor Data Mover: 2-D f16 tile (rows x cols) global -> LDS ----
// D# per CDNA5 ISA; tensor described as exactly the
// tile (no OOB), row stride = stride_elems, LDS written contiguously.
__device__ inline void tdm_load_tile_f16(unsigned lds_off, const half_t* g,
                                         unsigned rows, unsigned cols,
                                         unsigned stride_elems) {
    const unsigned long long ga = (unsigned long long)(uintptr_t)g;
    u32x4 g0 = {};
    g0[0] = 1u;                                            // count=1, user mode
    g0[1] = lds_off;                                       // lds_addr
    g0[2] = (unsigned)ga;                                  // global_addr[31:0]
    g0[3] = (unsigned)((ga >> 32) & 0x1FFFFFFull) | (2u << 30); // addr[56:32] | type=2
    u32x8 g1 = {};
    g1[0] = (1u << 16);                                    // data_size = 2B
    g1[1] = (cols & 0xFFFFu) << 16;                        // tensor_dim0[15:0]
    g1[2] = ((cols >> 16) & 0xFFFFu) | ((rows & 0xFFFFu) << 16); // t_dim0 hi | t_dim1 lo
    g1[3] = ((rows >> 16) & 0xFFFFu) | ((cols & 0xFFFFu) << 16); // t_dim1 hi | tile_dim0
    g1[4] = rows & 0xFFFFu;                                // tile_dim1 (tile_dim2=0)
    g1[5] = stride_elems;                                  // tensor_dim0_stride[31:0]
    g1[6] = 0;                                             // stride hi | dim1_stride lo
    g1[7] = 0;
    u32x4 g2 = {};
    u32x4 g3 = {};
    asm volatile("tensor_load_to_lds %0, %1, %2, %3"
                 :: "s"(g0), "s"(g1), "s"(g2), "s"(g3) : "memory");
}

// A fragment: 16x32 f16, row-major source, leading dim ld (halves).
// lane L -> row M=L&15; K = {0..7,16..23} (lanes 0-15) / {8..15,24..31} (lanes 16-31)
__device__ inline v16h load_a_frag(const half_t* base, int ld) {
    const int lane = threadIdx.x & 31;
    const int row  = lane & 15;
    const int kb   = (lane >> 4) << 3;
    const half_t* r = base + (size_t)row * ld;
    V16U u;
    u.h8[0] = *(const v8h*)(r + kb);
    u.h8[1] = *(const v8h*)(r + kb + 16);
    return u.v;
}

// B fragment: 32x16 (KxN); source stored N-major with K contiguous (base + n*ld + k).
// lane L -> col N=L&15; K = {0..7,16..23} (lanes 0-15) / {8..15,24..31} (lanes 16-31)
__device__ inline v16h load_b_frag(const half_t* base, int ld) {
    const int lane = threadIdx.x & 31;
    const int n = lane & 15;
    const int g = lane >> 4;
    const half_t* r = base + (size_t)n * ld + g * 8;
    V16U u;
    u.h8[0] = *(const v8h*)(r);
    u.h8[1] = *(const v8h*)(r + 16);
    return u.v;
}

// ---------------------------------------------------------------------------
// Prep: f32 -> f16 elementwise
// ---------------------------------------------------------------------------
__global__ __launch_bounds__(256) void cvt_f32_f16(
    const float* __restrict__ in, half_t* __restrict__ out, int n)
{
    int i = blockIdx.x * 256 + threadIdx.x;
    if (i < n) out[i] = (half_t)in[i];
}

// Prep: WT[n*K + k] = (f16) W[k*N + n]   (32x32 tiles via LDS)
__global__ __launch_bounds__(256) void tr_f32_f16(
    const float* __restrict__ W, half_t* __restrict__ WT, int K, int N)
{
    __shared__ float tl[32][33];
    const int tx = threadIdx.x, ty = threadIdx.y;    // block (32,8)
    const int n0 = blockIdx.x * 32, k0 = blockIdx.y * 32;
#pragma unroll
    for (int j = 0; j < 4; ++j)
        tl[ty + 8 * j][tx] = W[(size_t)(k0 + ty + 8 * j) * N + n0 + tx];
    __syncthreads();
#pragma unroll
    for (int j = 0; j < 4; ++j)
        WT[(size_t)(n0 + ty + 8 * j) * K + k0 + tx] = (half_t)tl[tx][ty + 8 * j];
}

// ---------------------------------------------------------------------------
// GEMM: C[M,N] = f16A[M,K] @ f16Bt[N,K]^T (+bias[col]).
// 256 threads = 8 waves; block tile 128x128; per-wave 32x64; k-step 32.
// A tile staged by the Tensor Data Mover (one wave-level DMA, TENSORcnt);
// Bt tile staged by per-lane async global->LDS b128 copies (ASYNCcnt).
// outmode: 0 = f16 row-major, 1 = f16 scatter to VT[(row&7)*1024+col][row>>3],
//          2 = f32 row-major.
// ---------------------------------------------------------------------------
__global__ __launch_bounds__(256) void gemm_tf16(
    const half_t* __restrict__ A, const half_t* __restrict__ Bt,
    const float* __restrict__ bias,
    half_t* __restrict__ Ch, float* __restrict__ Cf,
    int ldc, int K, int outmode)
{
    __shared__ __align__(16) half_t As[128][32];
    __shared__ __align__(16) half_t Bs[128][32];
    const int t    = threadIdx.x;
    const int wave = t >> 5;
    const int wr   = wave >> 1;          // 0..3  (32-row strip)
    const int wc   = wave & 1;           // 0..1  (64-col strip)
    const int n0g  = blockIdx.x * 128;
    const int m0g  = blockIdx.y * 128;

    const unsigned asBase = (unsigned)(uintptr_t)(&As[0][0]);
    const unsigned bsBase = (unsigned)(uintptr_t)(&Bs[0][0]);

    v8f acc[2][4] = {};

    for (int k0 = 0; k0 < K; k0 += 32) {
        // stage Bt[128][32]: 512 x 16B chunks, 2 per thread (ASYNCcnt)
#pragma unroll
        for (int j = 0; j < 2; ++j) {
            const int c   = t + 256 * j;
            const int row = c >> 2;
            const int prt = c & 3;
            async_copy16(bsBase + (unsigned)(row * 64 + prt * 16),
                         Bt + (size_t)(n0g + row) * K + k0 + prt * 8);
        }
        // stage As[128][32] with one TDM descriptor (TENSORcnt), wave 0 only
        if (wave == 0)
            tdm_load_tile_f16(asBase, A + (size_t)m0g * K + k0,
                              128u, 32u, (unsigned)K);
        async_wait0();
        if (wave == 0)
            __builtin_amdgcn_s_wait_tensorcnt(0);
        __syncthreads();

        v16h af[2];
#pragma unroll
        for (int mt = 0; mt < 2; ++mt)
            af[mt] = load_a_frag(&As[wr * 32 + mt * 16][0], 32);
#pragma unroll
        for (int nt = 0; nt < 4; ++nt) {
            v16h bf = load_b_frag(&Bs[wc * 64 + nt * 16][0], 32);
#pragma unroll
            for (int mt = 0; mt < 2; ++mt)
                acc[mt][nt] = WMMA_F16(af[mt], bf, acc[mt][nt]);
        }
        __syncthreads();
    }

    const int lane   = t & 31;
    const int rowoff = (lane >> 4) * 8;
#pragma unroll
    for (int mt = 0; mt < 2; ++mt)
#pragma unroll
        for (int nt = 0; nt < 4; ++nt) {
            const int gcol = n0g + wc * 64 + nt * 16 + (lane & 15);
            const float bv = bias ? bias[gcol] : 0.f;
#pragma unroll
            for (int v = 0; v < 8; ++v) {
                const int grow = m0g + wr * 32 + mt * 16 + rowoff + v;
                const float r = acc[mt][nt][v] + bv;
                if (outmode == 0) {
                    Ch[(size_t)grow * ldc + gcol] = (half_t)r;
                } else if (outmode == 1) {
                    // V rows are (i*8+b); scatter to VT[(b*1024 + gcol)][i]
                    Ch[(size_t)((grow & 7) * 1024 + gcol) * 1024 + (grow >> 3)] = (half_t)r;
                } else {
                    Cf[(size_t)grow * ldc + gcol] = r;
                }
            }
        }
}

// ---------------------------------------------------------------------------
// Relative attention, flash-style streaming softmax. No block barriers:
// each wave owns 16 query rows; V is pre-transposed in global (VT).
// Grid: (qblock=16, batch=8, head=16); block = 64 threads = 2 waves.
// rel_shift: attn_p[q,i] = rawp[q-q0][i+511-q] (always in-bounds when unmasked)
// ---------------------------------------------------------------------------
__global__ __launch_bounds__(64) void relattn_kernel(
    const half_t* __restrict__ QU, const half_t* __restrict__ QV,
    const half_t* __restrict__ Kb, const half_t* __restrict__ VT,
    const half_t* __restrict__ Pb, half_t* __restrict__ ATT)
{
    extern __shared__ char smem[];
    half_t* rawp  = (half_t*)smem;                   // [32][1024]
    half_t* probs = (half_t*)(smem + 32 * 1024 * 2); // [2][16][32]

    const int t      = threadIdx.x;
    const int lane   = t & 31;
    const int wave   = t >> 5;
    const int q0     = blockIdx.x * 32;
    const int b      = blockIdx.y;
    const int h      = blockIdx.z;
    const int qrow0  = q0 + wave * 16;
    const int rowoff = (lane >> 4) * 8;
    const int coln   = lane & 15;

    const size_t qoff = ((size_t)qrow0 * 8 + b) * 1024 + (size_t)h * 64;
    v16h a_qu[2], a_qv[2];
#pragma unroll
    for (int ks = 0; ks < 2; ++ks) {
        a_qu[ks] = load_a_frag(QU + qoff + ks * 32, 8 * 1024);
        a_qv[ks] = load_a_frag(QV + qoff + ks * 32, 8 * 1024);
    }

    // Phase A: raw positional scores (q+v) @ P^T for this wave's rows
    for (int jt = 0; jt < 64; ++jt) {
        v8f acc = {};
#pragma unroll
        for (int ks = 0; ks < 2; ++ks) {
            v16h bf = load_b_frag(Pb + (size_t)(jt * 16) * 1024 + h * 64 + ks * 32, 1024);
            acc = WMMA_F16(a_qv[ks], bf, acc);
        }
        const int n = jt * 16 + coln;
#pragma unroll
        for (int v = 0; v < 8; ++v)
            rawp[(wave * 16 + rowoff + v) * 1024 + n] = (half_t)acc[v];
    }

    float m_i[8], l_i[8];
    v8f o_acc[4] = {};
#pragma unroll
    for (int v = 0; v < 8; ++v) { m_i[v] = -__builtin_inff(); l_i[v] = 0.f; }

    const half_t* vtb = VT + (size_t)(b * 1024 + h * 64) * 1024; // [64 d][1024 i]
    const int nkey = q0 + 544;                   // mask horizon (multiple of 32)
    for (int i0 = 0; i0 < nkey; i0 += 32) {
        // prefetch next key/value tiles into cache
        if (i0 + 32 < nkey) {
            __builtin_prefetch(Kb + ((size_t)(i0 + 32) * 8 + b) * 1024 + h * 64, 0, 0);
            __builtin_prefetch(vtb + (size_t)coln * 1024 + i0 + 32, 0, 0);
        }

        // content scores: (q+u) @ K^T, two 16-key tiles
        v8f s0 = {}, s1 = {};
#pragma unroll
        for (int ks = 0; ks < 2; ++ks) {
            v16h bk0 = load_b_frag(Kb + ((size_t)(i0)      * 8 + b) * 1024 + h * 64 + ks * 32, 8 * 1024);
            v16h bk1 = load_b_frag(Kb + ((size_t)(i0 + 16) * 8 + b) * 1024 + h * 64 + ks * 32, 8 * 1024);
            s0 = WMMA_F16(a_qu[ks], bk0, s0);
            s1 = WMMA_F16(a_qu[ks], bk1, s1);
        }

        // add shifted positional scores, mask, scale
        float val0[8], val1[8];
#pragma unroll
        for (int v = 0; v < 8; ++v) {
            const int q  = qrow0 + rowoff + v;
            const int ia = i0 + coln;
            const int ib = i0 + 16 + coln;
            val0[v] = (ia > q + 512) ? -__builtin_inff()
                    : (s0[v] + (float)rawp[(q - q0) * 1024 + (ia + 511 - q)]) * SCALE_F;
            val1[v] = (ib > q + 512) ? -__builtin_inff()
                    : (s1[v] + (float)rawp[(q - q0) * 1024 + (ib + 511 - q)]) * SCALE_F;
        }

        // row max across the 16-lane half-wave
        float cand[8];
#pragma unroll
        for (int v = 0; v < 8; ++v) cand[v] = fmaxf(val0[v], val1[v]);
#pragma unroll
        for (int off = 1; off < 16; off <<= 1)
#pragma unroll
            for (int v = 0; v < 8; ++v)
                cand[v] = fmaxf(cand[v], __shfl_xor(cand[v], off, 32));

        float p0[8], p1[8], rs[8], corr[8];
#pragma unroll
        for (int v = 0; v < 8; ++v) {
            const float mn = fmaxf(m_i[v], cand[v]);
            corr[v] = __expf(m_i[v] - mn);
            p0[v]   = __expf(val0[v] - mn);
            p1[v]   = __expf(val1[v] - mn);
            rs[v]   = p0[v] + p1[v];
            m_i[v]  = mn;
        }
#pragma unroll
        for (int off = 1; off < 16; off <<= 1)
#pragma unroll
            for (int v = 0; v < 8; ++v)
                rs[v] += __shfl_xor(rs[v], off, 32);
#pragma unroll
        for (int v = 0; v < 8; ++v) l_i[v] = l_i[v] * corr[v] + rs[v];
#pragma unroll
        for (int nt = 0; nt < 4; ++nt)
#pragma unroll
            for (int v = 0; v < 8; ++v)
                o_acc[nt][v] *= corr[v];

        // probabilities: C-layout -> A-layout via per-wave LDS region
        half_t* pw = probs + wave * (16 * 32);
#pragma unroll
        for (int v = 0; v < 8; ++v) {
            pw[(rowoff + v) * 32 + coln]      = (half_t)p0[v];
            pw[(rowoff + v) * 32 + 16 + coln] = (half_t)p1[v];
        }
        v16h a_p = load_a_frag(pw, 32);
#pragma unroll
        for (int nt = 0; nt < 4; ++nt) {
            v16h bv = load_b_frag(vtb + (size_t)(nt * 16) * 1024 + i0, 1024);
            o_acc[nt] = WMMA_F16(a_p, bv, o_acc[nt]);
        }
    }

    // normalize + store (f16)
#pragma unroll
    for (int nt = 0; nt < 4; ++nt)
#pragma unroll
        for (int v = 0; v < 8; ++v) {
            const float o = o_acc[nt][v] / l_i[v];
            ATT[((size_t)(qrow0 + rowoff + v) * 8 + b) * 1024 + h * 64 + nt * 16 + coln] = (half_t)o;
        }
}

// ---------------------------------------------------------------------------
extern "C" void kernel_launch(void* const* d_in, const int* in_sizes, int n_in,
                              void* d_out, int out_size, void* d_ws, size_t ws_size,
                              hipStream_t stream)
{
    (void)in_sizes; (void)n_in; (void)out_size; (void)ws_size;
    const float* emb_new = (const float*)d_in[0];
    const float* emb_old = (const float*)d_in[1];
    const float* emb_pos = (const float*)d_in[2];
    const float* u_      = (const float*)d_in[3];
    const float* v_      = (const float*)d_in[4];
    // d_in[5] = mask (recomputed analytically)
    const float* Wq      = (const float*)d_in[6];
    const float* Wkv     = (const float*)d_in[7];
    const float* Wp      = (const float*)d_in[8];
    const float* Wfc     = (const float*)d_in[9];
    float* out = (float*)d_out;

    const size_t MB = 1024 * 1024;
    char* ws = (char*)d_ws;
    half_t* EMBH = (half_t*)(ws);            // 16 MiB [8192,1024] concat(old,new)
    half_t* POSH = (half_t*)(ws + 16 * MB);  //  2 MiB [1024,1024]
    half_t* WqT  = (half_t*)(ws + 18 * MB);  //  2 MiB [1024,1024]
    half_t* WkvT = (half_t*)(ws + 20 * MB);  //  4 MiB [2048,1024]
    half_t* WpT  = (half_t*)(ws + 24 * MB);  //  2 MiB
    half_t* WfcT = (half_t*)(ws + 26 * MB);  //  2 MiB
    half_t* QU   = (half_t*)(ws + 28 * MB);  //  8 MiB [4096,1024]
    half_t* QV   = (half_t*)(ws + 36 * MB);  //  8 MiB
    half_t* Kb   = (half_t*)(ws + 44 * MB);  // 16 MiB [8192,1024]
    half_t* VT   = (half_t*)(ws + 60 * MB);  // 16 MiB [8][16][64][1024]
    half_t* Pb   = (half_t*)(ws + 76 * MB);  //  2 MiB [1024,1024]
    half_t* ATT  = (half_t*)(ws + 78 * MB);  //  8 MiB [4096,1024]

    // ---- prep: conversions and weight transposes ----
    cvt_f32_f16<<<dim3(4096 * 1024 / 256), 256, 0, stream>>>(emb_old, EMBH, 4096 * 1024);
    cvt_f32_f16<<<dim3(4096 * 1024 / 256), 256, 0, stream>>>(emb_new, EMBH + (size_t)4096 * 1024, 4096 * 1024);
    cvt_f32_f16<<<dim3(1024 * 1024 / 256), 256, 0, stream>>>(emb_pos, POSH, 1024 * 1024);
    tr_f32_f16<<<dim3(32, 32), dim3(32, 8), 0, stream>>>(Wq,  WqT,  1024, 1024);
    tr_f32_f16<<<dim3(64, 32), dim3(32, 8), 0, stream>>>(Wkv, WkvT, 1024, 2048);
    tr_f32_f16<<<dim3(32, 32), dim3(32, 8), 0, stream>>>(Wp,  WpT,  1024, 1024);
    tr_f32_f16<<<dim3(32, 32), dim3(32, 8), 0, stream>>>(Wfc, WfcT, 1024, 1024);

    // ---- projections (WMMA + TDM/async LDS staging) ----
    // K: [8192,1024]
    gemm_tf16<<<dim3(8, 64), 256, 0, stream>>>(EMBH, WkvT, nullptr,
                                               Kb, nullptr, 1024, 1024, 0);
    // V: scatter-transposed to VT
    gemm_tf16<<<dim3(8, 64), 256, 0, stream>>>(EMBH, WkvT + (size_t)1024 * 1024, nullptr,
                                               VT, nullptr, 1024, 1024, 1);
    // q+u, q+v
    gemm_tf16<<<dim3(8, 32), 256, 0, stream>>>(EMBH + (size_t)4096 * 1024, WqT, u_,
                                               QU, nullptr, 1024, 1024, 0);
    gemm_tf16<<<dim3(8, 32), 256, 0, stream>>>(EMBH + (size_t)4096 * 1024, WqT, v_,
                                               QV, nullptr, 1024, 1024, 0);
    // positional
    gemm_tf16<<<dim3(8, 8), 256, 0, stream>>>(POSH, WpT, nullptr,
                                              Pb, nullptr, 1024, 1024, 0);

    // ---- attention ----
    const size_t smem = (size_t)32 * 1024 * 2 + 2 * 16 * 32 * 2; // 67584 B
    relattn_kernel<<<dim3(16, 8, 16), 64, smem, stream>>>(QU, QV, Kb, VT, Pb, ATT);

    // ---- final projection -> f32 out ----
    gemm_tf16<<<dim3(8, 32), 256, 0, stream>>>(ATT, WfcT, nullptr,
                                               nullptr, out, 1024, 1024, 2);
}
